// IEGRLayer_33517924778682
// MI455X (gfx1250) — compile-verified, matrix operand order checked by reading, not run on hardware
//
#include <hip/hip_runtime.h>
#include <hip/hip_bf16.h>
#include <stdint.h>

#define HID 128
#define NIN 64
#define EIN 32
#define OUTD 128
#define WAVES 2

typedef __attribute__((ext_vector_type(16))) __bf16 v16bf;
typedef __attribute__((ext_vector_type(8)))  float  f32x8;

union Frag { v16bf v; uint32_t u[8]; };

static __device__ __forceinline__ uint16_t f2bf(float f) {
    uint32_t x = __float_as_uint(f);
    uint32_t r = x + 0x7FFFu + ((x >> 16) & 1u);
    return (uint16_t)(r >> 16);
}
static __device__ __forceinline__ float lrelu(float v) {
    return v > 0.0f ? v : 0.01f * v;
}
static __device__ __forceinline__ int imin(int a, int b) { return a < b ? a : b; }

// ---------------------------------------------------------------------------
// Zero a float region (aggregation buffers), grid-stride.
// ---------------------------------------------------------------------------
__global__ void zero_f32(float* __restrict__ p, int n) {
    for (int i = blockIdx.x * blockDim.x + threadIdx.x; i < n;
         i += gridDim.x * blockDim.x)
        p[i] = 0.0f;
}

// ---------------------------------------------------------------------------
// Repack a (Kreal x 128) row-major f32 weight into bf16 B-fragments laid out
// per CDNA5 WMMA 16-bit B layout. One dword = packed (k, k+1) pair for
// lane l, vgpr v of tile (kt, nt):
//   n = nt*16 + (l&15), half = l>>4, k = kt*32 + 2*(v&3) + 16*(v>>2) + 8*half
// Storage: dst[((kt*8+nt)*32 + l)*8 + v]
// ---------------------------------------------------------------------------
__global__ void pack_w(const float* __restrict__ W, uint32_t* __restrict__ dst,
                       int Kreal, int Ktiles) {
    int total = Ktiles * 2048;
    for (int t = blockIdx.x * blockDim.x + threadIdx.x; t < total;
         t += gridDim.x * blockDim.x) {
        int v  = t & 7;
        int l  = (t >> 3) & 31;
        int nt = (t >> 8) & 7;
        int kt = t >> 11;
        int n  = nt * 16 + (l & 15);
        int lh = l >> 4;
        int k  = kt * 32 + 2 * (v & 3) + 16 * (v >> 2) + 8 * lh;
        float f0 = (k     < Kreal) ? W[(size_t)k       * 128 + n] : 0.0f;
        float f1 = (k + 1 < Kreal) ? W[(size_t)(k + 1) * 128 + n] : 0.0f;
        dst[t] = (uint32_t)f2bf(f0) | ((uint32_t)f2bf(f1) << 16);
    }
}

// ---------------------------------------------------------------------------
// Edge kernel: 1 wave = 16 edges. Per-wave LDS (uint32 units):
//   [0,2560)     e_in bf16 16x320   (reused as t f32 16x128 for coords stage)
//   [2560,4608)  h1   f32  16x128
//   [4608,5632)  A    bf16 16x128   (h1_ln then m_ij)
//   [5632,5680)  x_rel f32 16x3
//   [5680,5712)  src/dst ids 2x16
// ---------------------------------------------------------------------------
#define EW_DW 5712
__launch_bounds__(WAVES * 32)
__global__ void edge_kernel(const float* __restrict__ feats,
                            const float* __restrict__ x,
                            const float* __restrict__ edge_feats,
                            const int* __restrict__ src,
                            const int* __restrict__ dst,
                            const float* __restrict__ e_b1,
                            const float* __restrict__ e_g1,
                            const float* __restrict__ e_bn1,
                            const float* __restrict__ e_b2,
                            const float* __restrict__ c_b1,
                            const float* __restrict__ c_W2,
                            const uint32_t* __restrict__ wE1,
                            const uint32_t* __restrict__ wE2,
                            const uint32_t* __restrict__ wC1,
                            float* __restrict__ aggr,
                            float* __restrict__ x_sum,
                            float* __restrict__ cnt,
                            int E) {
    __shared__ uint32_t smem[WAVES * EW_DW];
    const int wave = threadIdx.x >> 5;
    const int lane = threadIdx.x & 31;
    const int mm   = lane & 15;   // column group / row-in-tile
    const int lh   = lane >> 4;   // lane half

    uint32_t* base = smem + wave * EW_DW;
    uint16_t* einH = (uint16_t*)base;             // bf16 16x320
    uint32_t* einU = base;
    float*    h1f  = (float*)(base + 2560);       // f32 16x128
    uint16_t* abf  = (uint16_t*)(base + 4608);    // bf16 16x128
    const uint32_t* abfU = base + 4608;
    float*    tf   = (float*)base;                // reuse e_in region
    float*    xrel = (float*)(base + 5632);       // 16x3
    int*      sd   = (int*)(base + 5680);         // src[16], dst[16]

    const int tile = blockIdx.x * WAVES + wave;
    const int eb   = tile * 16;

    // ---- per-edge geometry + RBF (lanes 0..15, one edge each) ----
    if (lane < 16) {
        int e = imin(eb + lane, E - 1);
        int s = src[e], d = dst[e];
        sd[lane]      = s;
        sd[16 + lane] = d;
        float dx = x[3 * s + 0] - x[3 * d + 0];
        float dy = x[3 * s + 1] - x[3 * d + 1];
        float dz = x[3 * s + 2] - x[3 * d + 2];
        float nrm = sqrtf(dx * dx + dy * dy + dz * dz) + 1.0f;
        dx /= nrm; dy /= nrm; dz /= nrm;
        xrel[3 * lane + 0] = dx;
        xrel[3 * lane + 1] = dy;
        xrel[3 * lane + 2] = dz;
        float d2 = dx * dx + dy * dy + dz * dz;
        float sg = 1.0f;
        for (int k = 0; k < 15; ++k) {
            einH[lane * 320 + 288 + k] = f2bf(expf(-d2 / sg));
            sg *= 1.5f;
        }
        for (int k = 15; k < 32; ++k) einH[lane * 320 + 288 + k] = 0;  // pad K->320
    }
    __syncthreads();

    // ---- gather node + edge features into e_in rows (bf16) ----
    for (int r = 0; r < 16; ++r) {
        int s = sd[r], d = sd[16 + r];
        for (int j = lane; j < 128; j += 32) {
            einH[r * 320 + j]       = f2bf(feats[(size_t)s * 128 + j]);
            einH[r * 320 + 128 + j] = f2bf(feats[(size_t)d * 128 + j]);
        }
        int e = imin(eb + r, E - 1);
        einH[r * 320 + 256 + lane] = f2bf(edge_feats[(size_t)e * 32 + lane]);
    }
    __syncthreads();

    // ---- stage 1: h1 = lrelu(e_in @ e_W1 + b1) ----
    for (int nt = 0; nt < 8; ++nt) {
        f32x8 a = {0.f, 0.f, 0.f, 0.f, 0.f, 0.f, 0.f, 0.f};
        for (int kt = 0; kt < 10; ++kt) {
            Frag fa, fb;
#pragma unroll
            for (int v = 0; v < 8; ++v) {
                int k = kt * 32 + 2 * (v & 3) + 16 * (v >> 2) + 8 * lh;
                fa.u[v] = einU[(mm * 320 + k) >> 1];
                fb.u[v] = wE1[((size_t)((kt * 8 + nt) * 32 + lane)) * 8 + v];
            }
            a = __builtin_amdgcn_wmma_f32_16x16x32_bf16(
                    false, fa.v, false, fb.v, (short)0, a, false, false);
        }
        int n = nt * 16 + mm;
        float b = e_b1[n];
#pragma unroll
        for (int r = 0; r < 8; ++r)
            h1f[(r + 8 * lh) * 128 + n] = lrelu(a[r] + b);
    }
    __syncthreads();

    // ---- LayerNorm h1 -> bf16 A ----
    if (lane < 16) {
        float s1 = 0.f, s2 = 0.f;
        for (int j = 0; j < 128; ++j) {
            float v = h1f[lane * 128 + j];
            s1 += v; s2 += v * v;
        }
        float mu  = s1 * (1.0f / 128.0f);
        float var = s2 * (1.0f / 128.0f) - mu * mu;
        float inv = rsqrtf(var + 1e-5f);
        for (int j = 0; j < 128; ++j) {
            float v = (h1f[lane * 128 + j] - mu) * inv * e_g1[j] + e_bn1[j];
            abf[lane * 128 + j] = f2bf(v);
        }
    }
    __syncthreads();

    // ---- stage 2: m_ij = lrelu(h1_ln @ e_W2 + b2); scatter into aggr ----
    for (int nt = 0; nt < 8; ++nt) {
        f32x8 a = {0.f, 0.f, 0.f, 0.f, 0.f, 0.f, 0.f, 0.f};
        for (int kt = 0; kt < 4; ++kt) {
            Frag fa, fb;
#pragma unroll
            for (int v = 0; v < 8; ++v) {
                int k = kt * 32 + 2 * (v & 3) + 16 * (v >> 2) + 8 * lh;
                fa.u[v] = abfU[(mm * 128 + k) >> 1];
                fb.u[v] = wE2[((size_t)((kt * 8 + nt) * 32 + lane)) * 8 + v];
            }
            a = __builtin_amdgcn_wmma_f32_16x16x32_bf16(
                    false, fa.v, false, fb.v, (short)0, a, false, false);
        }
        int n = nt * 16 + mm;
        float b = e_b2[n];
#pragma unroll
        for (int r = 0; r < 8; ++r) {
            int row = r + 8 * lh;
            float v = lrelu(a[r] + b);
            h1f[row * 128 + n] = v;   // stash m_ij (f32)
            if (eb + row < E)
                atomicAdd(&aggr[(size_t)sd[16 + row] * 128 + n], v);
        }
    }
    __syncthreads();

    // ---- m_ij f32 -> bf16 A for coords MLP ----
    for (int idx = lane; idx < 2048; idx += 32) abf[idx] = f2bf(h1f[idx]);
    __syncthreads();

    // ---- stage 3: t = lrelu(m_ij @ c_W1 + c_b1) ----
    for (int nt = 0; nt < 8; ++nt) {
        f32x8 a = {0.f, 0.f, 0.f, 0.f, 0.f, 0.f, 0.f, 0.f};
        for (int kt = 0; kt < 4; ++kt) {
            Frag fa, fb;
#pragma unroll
            for (int v = 0; v < 8; ++v) {
                int k = kt * 32 + 2 * (v & 3) + 16 * (v >> 2) + 8 * lh;
                fa.u[v] = abfU[(mm * 128 + k) >> 1];
                fb.u[v] = wC1[((size_t)((kt * 8 + nt) * 32 + lane)) * 8 + v];
            }
            a = __builtin_amdgcn_wmma_f32_16x16x32_bf16(
                    false, fa.v, false, fb.v, (short)0, a, false, false);
        }
        int n = nt * 16 + mm;
        float b = c_b1[n];
#pragma unroll
        for (int r = 0; r < 8; ++r)
            tf[(r + 8 * lh) * 128 + n] = lrelu(a[r] + b);
    }
    __syncthreads();

    // ---- c = t . c_W2 ; coordinate scatter ----
    if (lane < 16 && (eb + lane) < E) {
        float c = 0.f;
        for (int j = 0; j < 128; ++j) c += tf[lane * 128 + j] * c_W2[j];
        int d = sd[16 + lane];
        atomicAdd(&x_sum[d * 3 + 0], xrel[3 * lane + 0] * c);
        atomicAdd(&x_sum[d * 3 + 1], xrel[3 * lane + 1] * c);
        atomicAdd(&x_sum[d * 3 + 2], xrel[3 * lane + 2] * c);
        atomicAdd(&cnt[d], 1.0f);
    }
}

// ---------------------------------------------------------------------------
// Node kernel: 1 wave = 16 nodes. Per-wave LDS (uint32 units):
//   [0,3584)     n_in bf16 16x448
//   [3584,5632)  h    f32  16x128
//   [5632,6656)  A    bf16 16x128
// ---------------------------------------------------------------------------
#define NW_DW 6656
__launch_bounds__(WAVES * 32)
__global__ void node_kernel(const float* __restrict__ feats,
                            const float* __restrict__ orig,
                            const float* __restrict__ x,
                            const float* __restrict__ nn_g,
                            const float* __restrict__ nn_b,
                            const float* __restrict__ n_b1,
                            const float* __restrict__ n_g1,
                            const float* __restrict__ n_bn1,
                            const float* __restrict__ n_b2,
                            const uint32_t* __restrict__ wN1,
                            const uint32_t* __restrict__ wN2,
                            const float* __restrict__ aggr,
                            const float* __restrict__ x_sum,
                            const float* __restrict__ cnt,
                            float* __restrict__ out,
                            int N) {
    __shared__ uint32_t smem[WAVES * NW_DW];
    const int wave = threadIdx.x >> 5;
    const int lane = threadIdx.x & 31;
    const int mm   = lane & 15;
    const int lh   = lane >> 4;

    uint32_t* base = smem + wave * NW_DW;
    uint16_t* ninH = (uint16_t*)base;            // bf16 16x448
    uint32_t* ninU = base;
    float*    hf   = (float*)(base + 3584);      // f32 16x128
    uint16_t* abf  = (uint16_t*)(base + 5632);   // bf16 16x128
    const uint32_t* abfU = base + 5632;

    const int tile = blockIdx.x * WAVES + wave;
    const int nb   = tile * 16;

    // ---- cols 0..127: LN(feats) per row ----
    if (lane < 16) {
        int i = imin(nb + lane, N - 1);
        float s1 = 0.f, s2 = 0.f;
        for (int j = 0; j < 128; ++j) {
            float v = feats[(size_t)i * 128 + j];
            s1 += v; s2 += v * v;
        }
        float mu  = s1 * (1.0f / 128.0f);
        float var = s2 * (1.0f / 128.0f) - mu * mu;
        float inv = rsqrtf(var + 1e-5f);
        for (int j = 0; j < 128; ++j) {
            float v = (feats[(size_t)i * 128 + j] - mu) * inv * nn_g[j] + nn_b[j];
            ninH[lane * 448 + j] = f2bf(v);
        }
    }
    // ---- cols 128..255: aggr_msg; 256..383: zeros; 384..447: orig feats ----
    for (int r = 0; r < 16; ++r) {
        int i = imin(nb + r, N - 1);
        for (int j = lane; j < 128; j += 32) {
            ninH[r * 448 + 128 + j] = f2bf(aggr[(size_t)i * 128 + j]);
            ninH[r * 448 + 256 + j] = 0;
        }
        for (int j = lane; j < 64; j += 32)
            ninH[r * 448 + 384 + j] = f2bf(orig[(size_t)i * 64 + j]);
    }
    __syncthreads();

    // ---- stage 1: h = lrelu(n_in @ n_W1 + b1) ----
    for (int nt = 0; nt < 8; ++nt) {
        f32x8 a = {0.f, 0.f, 0.f, 0.f, 0.f, 0.f, 0.f, 0.f};
        for (int kt = 0; kt < 14; ++kt) {
            Frag fa, fb;
#pragma unroll
            for (int v = 0; v < 8; ++v) {
                int k = kt * 32 + 2 * (v & 3) + 16 * (v >> 2) + 8 * lh;
                fa.u[v] = ninU[(mm * 448 + k) >> 1];
                fb.u[v] = wN1[((size_t)((kt * 8 + nt) * 32 + lane)) * 8 + v];
            }
            a = __builtin_amdgcn_wmma_f32_16x16x32_bf16(
                    false, fa.v, false, fb.v, (short)0, a, false, false);
        }
        int n = nt * 16 + mm;
        float b = n_b1[n];
#pragma unroll
        for (int r = 0; r < 8; ++r)
            hf[(r + 8 * lh) * 128 + n] = lrelu(a[r] + b);
    }
    __syncthreads();

    // ---- LayerNorm h -> bf16 A ----
    if (lane < 16) {
        float s1 = 0.f, s2 = 0.f;
        for (int j = 0; j < 128; ++j) {
            float v = hf[lane * 128 + j];
            s1 += v; s2 += v * v;
        }
        float mu  = s1 * (1.0f / 128.0f);
        float var = s2 * (1.0f / 128.0f) - mu * mu;
        float inv = rsqrtf(var + 1e-5f);
        for (int j = 0; j < 128; ++j) {
            float v = (hf[lane * 128 + j] - mu) * inv * n_g1[j] + n_bn1[j];
            abf[lane * 128 + j] = f2bf(v);
        }
    }
    __syncthreads();

    // ---- stage 2: node_out = h_ln @ n_W2 + b2 ; skip/mix; store ----
    for (int nt = 0; nt < 8; ++nt) {
        f32x8 a = {0.f, 0.f, 0.f, 0.f, 0.f, 0.f, 0.f, 0.f};
        for (int kt = 0; kt < 4; ++kt) {
            Frag fa, fb;
#pragma unroll
            for (int v = 0; v < 8; ++v) {
                int k = kt * 32 + 2 * (v & 3) + 16 * (v >> 2) + 8 * lh;
                fa.u[v] = abfU[(mm * 128 + k) >> 1];
                fb.u[v] = wN2[((size_t)((kt * 8 + nt) * 32 + lane)) * 8 + v];
            }
            a = __builtin_amdgcn_wmma_f32_16x16x32_bf16(
                    false, fa.v, false, fb.v, (short)0, a, false, false);
        }
        int n = nt * 16 + mm;
        float b = n_b2[n];
#pragma unroll
        for (int r = 0; r < 8; ++r) {
            int row = r + 8 * lh;
            int i = nb + row;
            if (i < N) {
                float no = a[r] + b;
                out[(size_t)i * 128 + n] =
                    0.75f * no + 0.25f * feats[(size_t)i * 128 + n];
            }
        }
    }

    // ---- x_new = x + x_sum / max(cnt,1) ----
    if (lane < 16) {
        int i = nb + lane;
        if (i < N) {
            float den = fmaxf(cnt[i], 1.0f);
            float inv = 1.0f / den;
#pragma unroll
            for (int k = 0; k < 3; ++k)
                out[(size_t)N * 128 + i * 3 + k] =
                    x[i * 3 + k] + x_sum[i * 3 + k] * inv;
        }
    }
}

// ---------------------------------------------------------------------------
extern "C" void kernel_launch(void* const* d_in, const int* in_sizes, int n_in,
                              void* d_out, int out_size, void* d_ws, size_t ws_size,
                              hipStream_t stream) {
    const float* feats      = (const float*)d_in[0];
    const float* orig       = (const float*)d_in[1];
    const float* x          = (const float*)d_in[2];
    // d_in[3] = x_orig unused (X_CONN_INIT == 0)
    const float* edge_feats = (const float*)d_in[4];
    const int*   src        = (const int*)d_in[5];
    const int*   dst        = (const int*)d_in[6];
    const float* e_W1  = (const float*)d_in[7];
    const float* e_b1  = (const float*)d_in[8];
    const float* e_g1  = (const float*)d_in[9];
    const float* e_bn1 = (const float*)d_in[10];
    const float* e_W2  = (const float*)d_in[11];
    const float* e_b2  = (const float*)d_in[12];
    const float* nn_g  = (const float*)d_in[13];
    const float* nn_b  = (const float*)d_in[14];
    const float* n_W1  = (const float*)d_in[15];
    const float* n_b1  = (const float*)d_in[16];
    const float* n_g1  = (const float*)d_in[17];
    const float* n_bn1 = (const float*)d_in[18];
    const float* n_W2  = (const float*)d_in[19];
    const float* n_b2  = (const float*)d_in[20];
    const float* c_W1  = (const float*)d_in[21];
    const float* c_b1  = (const float*)d_in[22];
    const float* c_W2  = (const float*)d_in[23];

    const int N = in_sizes[0] / HID;        // 50000
    const int E = in_sizes[5];              // 800000

    // Workspace layout (dwords)
    uint32_t* ws  = (uint32_t*)d_ws;
    uint32_t* wE1 = ws;                     // 10 ktiles * 2048
    uint32_t* wE2 = ws + 20480;             //  4 ktiles
    uint32_t* wC1 = ws + 28672;             //  4 ktiles
    uint32_t* wN1 = ws + 36864;             // 14 ktiles
    uint32_t* wN2 = ws + 65536;             //  4 ktiles
    float* fws    = (float*)(ws + 73728);
    float* aggr   = fws;                    // N*128
    float* x_sum  = fws + (size_t)N * 128;  // N*3
    float* cnt    = x_sum + (size_t)N * 3;  // N

    // Zero aggregation buffers (contiguous region)
    zero_f32<<<2048, 256, 0, stream>>>(fws, N * (128 + 3 + 1));

    // Pack weights into WMMA B-fragment layout
    pack_w<<<80, 256, 0, stream>>>(e_W1, wE1, 303, 10);
    pack_w<<<32, 256, 0, stream>>>(e_W2, wE2, 128, 4);
    pack_w<<<32, 256, 0, stream>>>(c_W1, wC1, 128, 4);
    pack_w<<<112, 256, 0, stream>>>(n_W1, wN1, 448, 14);
    pack_w<<<32, 256, 0, stream>>>(n_W2, wN2, 128, 4);

    const int Etiles = (E + 15) / 16;
    const int Ntiles = (N + 15) / 16;

    edge_kernel<<<(Etiles + WAVES - 1) / WAVES, WAVES * 32, 0, stream>>>(
        feats, x, edge_feats, src, dst, e_b1, e_g1, e_bn1, e_b2, c_b1, c_W2,
        wE1, wE2, wC1, aggr, x_sum, cnt, E);

    node_kernel<<<(Ntiles + WAVES - 1) / WAVES, WAVES * 32, 0, stream>>>(
        feats, orig, x, nn_g, nn_b, n_b1, n_g1, n_bn1, n_b2,
        wN1, wN2, aggr, x_sum, cnt, (float*)d_out, N);
}
